// KVMemnn_6141803233888
// MI455X (gfx1250) — matrix operand order repaired and checked by reading, not compiled, not run on hardware
//
#include <hip/hip_runtime.h>
#include <hip/hip_bf16.h>
#include <math.h>

// Problem constants (from reference)
#define VV 100000
#define DD 256
#define MM 4096
#define CC 4096
#define LL 32
#define M1 4097            // M + 1 (mems_enc rows incl. xs)
#define EPSF 1e-8f

typedef __attribute__((ext_vector_type(16))) _Float16 v16h;
typedef __attribute__((ext_vector_type(8)))  float    v8f;

// ---------------------------------------------------------------------------
// Kernel 1: freq-weighted embedding encode (the bandwidth-dominant kernel:
// ~256 MB of gathered 1KB embedding rows). 256-thread block = 4 token rows;
// each 64-thread sub-group owns one row, each lane owns 4 dims (float4 ->
// global_load_b128, 512B per wave-instruction). Per-row weight norm is a
// wave32 butterfly on lanes 0-31 of waves 0/2/4/6.
// ---------------------------------------------------------------------------
__global__ void encode_kernel(const int* __restrict__ toks,
                              const float* __restrict__ lt,
                              const float* __restrict__ freqs,
                              float* __restrict__ out, int nrows) {
    __shared__ int   s_tok[4][LL];
    __shared__ float s_w[4][LL];
    const int tid = threadIdx.x;
    const int rl  = tid >> 6;                 // row-local 0..3
    const int sub = tid & 63;                 // lane within sub-group
    const int row = blockIdx.x * 4 + rl;

    if (row < nrows && sub < LL) {            // full wave32s: waves 0,2,4,6
        int t = toks[(size_t)row * LL + sub];
        float f = freqs[t];
        s_tok[rl][sub] = t;
        float s = f * f;
        #pragma unroll
        for (int off = 16; off; off >>= 1) s += __shfl_xor(s, off, 32);
        s_w[rl][sub] = f / sqrtf(s);
    }
    __syncthreads();

    if (row < nrows) {
        float4 acc = make_float4(0.f, 0.f, 0.f, 0.f);
        #pragma unroll 4
        for (int l = 0; l < LL; ++l) {
            if (l + 4 < LL)
                __builtin_prefetch(lt + (size_t)s_tok[rl][l + 4] * DD + sub * 4, 0, 1);
            const float4 v = ((const float4*)(lt + (size_t)s_tok[rl][l] * DD))[sub];
            const float w = s_w[rl][l];
            acc.x += w * v.x; acc.y += w * v.y;
            acc.z += w * v.z; acc.w += w * v.w;
        }
        ((float4*)(out + (size_t)row * DD))[sub] = acc;
    }
}

// ---------------------------------------------------------------------------
// Kernel 2: dots[m] = <enc[m], enc[nrows-1]>  via v_wmma_f32_16x16x32_f16,
// plus bn[m] = ||enc[m]||_2. One wave (32 lanes) per 16 rows; K = 256 as
// 8 chunks of 32. A layout / B layout / C extraction per ISA 7.12.2 tables.
// ---------------------------------------------------------------------------
__global__ void dotnorm_kernel(const float* __restrict__ enc,
                               float* __restrict__ dots,
                               float* __restrict__ bn,
                               int nrows) {
    const int lane = threadIdx.x;            // 0..31
    const int base = blockIdx.x * 16;
    const int mrow = lane & 15;
    const int klo  = (lane < 16) ? 0 : 8;    // A-matrix low-K offset per lane half
    const int kbB  = (lane < 16) ? 0 : 16;   // unused; B uses klo, same k order as A

    long arow = base + mrow;                 // clamp so EXEC stays all-1s for WMMA
    if (arow > nrows - 1) arow = nrows - 1;
    const float* __restrict__ xsrow = enc + (size_t)(nrows - 1) * DD;
    const float* __restrict__ ap    = enc + (size_t)arow * DD;

    v8f c = {};
    #pragma unroll
    for (int ch = 0; ch < 8; ++ch) {
        const int k0 = ch * 32;
        v16h a, b;
        #pragma unroll
        for (int e = 0; e < 8; ++e) {
            a[e]     = (_Float16)ap[k0 + klo + e];        // VGPR0-3: K=klo+0..7
            a[e + 8] = (_Float16)ap[k0 + 16 + klo + e];   // VGPR4-7: K=16+klo+0..7
        }
        #pragma unroll
        for (int e = 0; e < 16; ++e)
            b[e] = (_Float16)xsrow[k0 + ((e & 8) << 1) + klo + (e & 7)];  // xs replicated in all 16 cols
        c = __builtin_amdgcn_wmma_f32_16x16x32_f16(
                false, a, false, b, (short)0, c, false, false);
    }

    // C layout: VGPR r -> (M=r, N=lane) for lanes 0-15, (M=r+8, N=lane-16) for 16-31.
    if (lane == 0) {
        #pragma unroll
        for (int r = 0; r < 8; ++r) { int m = base + r;     if (m < nrows) dots[m] = c[r]; }
    }
    if (lane == 16) {
        #pragma unroll
        for (int r = 0; r < 8; ++r) { int m = base + 8 + r; if (m < nrows) dots[m] = c[r]; }
    }

    // Row norms (f32, wave-reduced)
    for (int m = 0; m < 16; ++m) {
        int rowi = base + m;
        if (rowi >= nrows) break;
        const float* rp = enc + (size_t)rowi * DD;
        float s = 0.f;
        #pragma unroll
        for (int k = 0; k < 8; ++k) { float v = rp[lane + 32 * k]; s += v * v; }
        #pragma unroll
        for (int off = 16; off; off >>= 1) s += __shfl_xor(s, off, 32);
        if (lane == 0) bn[rowi] = sqrtf(s);
    }
}

// ---------------------------------------------------------------------------
// Kernel 3: cosine + softmax over 4097 rows (single block).
// ---------------------------------------------------------------------------
__global__ void softmax_kernel(const float* __restrict__ dots,
                               const float* __restrict__ bn,
                               float* __restrict__ att, int n) {
    __shared__ float red[256];
    const int tid = threadIdx.x;
    const float an = fmaxf(bn[n - 1], EPSF);

    float mx = -3.4e38f;
    for (int m = tid; m < n; m += 256) {
        float cv = dots[m] / (an * fmaxf(bn[m], EPSF));
        mx = fmaxf(mx, cv);
    }
    red[tid] = mx; __syncthreads();
    for (int s = 128; s; s >>= 1) { if (tid < s) red[tid] = fmaxf(red[tid], red[tid + s]); __syncthreads(); }
    mx = red[0]; __syncthreads();

    float sm = 0.f;
    for (int m = tid; m < n; m += 256) {
        float cv = dots[m] / (an * fmaxf(bn[m], EPSF));
        sm += __expf(cv - mx);
    }
    red[tid] = sm; __syncthreads();
    for (int s = 128; s; s >>= 1) { if (tid < s) red[tid] += red[tid + s]; __syncthreads(); }
    const float inv = 1.f / red[0];

    for (int m = tid; m < n; m += 256) {
        float cv = dots[m] / (an * fmaxf(bn[m], EPSF));
        att[m] = __expf(cv - mx) * inv;
    }
}

// ---------------------------------------------------------------------------
// Kernel 4: partial lhs[d] = sum_m att[m]*enc[m,d]; 32 deterministic partials.
// ---------------------------------------------------------------------------
__global__ void lhs_partial_kernel(const float* __restrict__ enc,
                                   const float* __restrict__ att,
                                   float* __restrict__ part, int n) {
    const int d = threadIdx.x;
    float acc = 0.f;
    for (int m = blockIdx.x; m < n; m += gridDim.x)
        acc += att[m] * enc[(size_t)m * DD + d];
    part[(size_t)blockIdx.x * DD + d] = acc;
}

__global__ void lhs_reduce_kernel(const float* __restrict__ part,
                                  float* __restrict__ lhs, int nblk) {
    const int d = threadIdx.x;
    float acc = 0.f;
    for (int b = 0; b < nblk; ++b) acc += part[(size_t)b * DD + d];
    lhs[d] = acc;
}

// ---------------------------------------------------------------------------
// Kernel 5: tile lhs into xs_out (4097 copies of the 256-vector), b128 stores.
// total4 = number of float4 elements (M1*DD/4). Stride multiples of 64 float4s
// keep each lane pinned to the same 4 dims.
// ---------------------------------------------------------------------------
__global__ void fill_kernel(const float* __restrict__ lhs,
                            float4* __restrict__ out4, size_t total4) {
    __shared__ float s[DD];
    s[threadIdx.x] = lhs[threadIdx.x];
    __syncthreads();
    const int g = (threadIdx.x & 63) * 4;
    const float4 v = make_float4(s[g], s[g + 1], s[g + 2], s[g + 3]);
    for (size_t i = (size_t)blockIdx.x * 256 + threadIdx.x; i < total4;
         i += (size_t)gridDim.x * 256)
        out4[i] = v;
}

// ---------------------------------------------------------------------------
extern "C" void kernel_launch(void* const* d_in, const int* in_sizes, int n_in,
                              void* d_out, int out_size, void* d_ws, size_t ws_size,
                              hipStream_t stream) {
    const int*   xs    = (const int*)d_in[0];
    const int*   mems  = (const int*)d_in[1];
    const int*   ys    = (const int*)d_in[2];
    const int*   cands = (const int*)d_in[3];
    const float* lt    = (const float*)d_in[4];
    const float* freqs = (const float*)d_in[5];
    float*       out   = (float*)d_out;

    // Workspace layout (floats): enc[M1*D] | dots[M1] | bn[M1] | att[M1] | lhs[D] | part[32*D]
    float* w    = (float*)d_ws;
    float* enc  = w;
    float* dots = enc  + (size_t)M1 * DD;
    float* bn   = dots + M1;
    float* att  = bn   + M1;
    float* lhs  = att  + M1;
    float* part = lhs  + DD;

    float* ys_out = out + (size_t)M1 * DD;   // second output half

    // Encodes (gather-bound; the bulk of the runtime). 4 rows per block.
    encode_kernel<<<(MM + 3) / 4, 256, 0, stream>>>(mems,  lt, freqs, enc, MM);
    encode_kernel<<<1,            256, 0, stream>>>(xs,    lt, freqs, enc + (size_t)MM * DD, 1);
    encode_kernel<<<1,            256, 0, stream>>>(ys,    lt, freqs, ys_out, 1);
    encode_kernel<<<(CC + 3) / 4, 256, 0, stream>>>(cands, lt, freqs, ys_out + DD, CC);

    // Attention (WMMA dots + norms, softmax, weighted sum)
    dotnorm_kernel<<<(M1 + 15) / 16, 32, 0, stream>>>(enc, dots, bn, M1);
    softmax_kernel<<<1, 256, 0, stream>>>(dots, bn, att, M1);
    lhs_partial_kernel<<<32, 256, 0, stream>>>(enc, att, part, M1);
    lhs_reduce_kernel<<<1, 256, 0, stream>>>(part, lhs, 32);

    // xs_out = lhs tiled 4097x (b128 stores)
    fill_kernel<<<1024, 256, 0, stream>>>(lhs, (float4*)out, ((size_t)M1 * DD) / 4);
}
